// OGBGCustomGINE_36283883716969
// MI455X (gfx1250) — matrix operand order repaired and checked by reading, not compiled, not run on hardware
//
#include <hip/hip_runtime.h>

#define N_NODES 40000
#define N_EDGES 640000
#define IN_DIM  128
#define EMB_DIM 256
#define LN_EPS  1e-5f

typedef _Float16 v16h __attribute__((ext_vector_type(16)));
typedef float    v8f  __attribute__((ext_vector_type(8)));

struct alignas(16) U4 { unsigned v[4]; };
union Frag { U4 u[2]; v16h h; };
struct alignas(8) H4 { _Float16 h[4]; };

// ---------------------------------------------------------------------------
// Zero the aggregation buffer (float4 stores)
// ---------------------------------------------------------------------------
__global__ __launch_bounds__(256) void zero_kernel(float4* __restrict__ p, int n4) {
    int i = blockIdx.x * 256 + threadIdx.x;
    if (i < n4) p[i] = make_float4(0.f, 0.f, 0.f, 0.f);
}

// ---------------------------------------------------------------------------
// Transpose + f32->f16 weight prep: Wt[n*K + k] = (f16) W[k*N + n]
// ---------------------------------------------------------------------------
__global__ __launch_bounds__(256) void wprep_kernel(const float* __restrict__ W,
                                                    _Float16* __restrict__ Wt,
                                                    int K, int N) {
    int idx = blockIdx.x * 256 + threadIdx.x;   // over N*K, contiguous in Wt
    if (idx >= N * K) return;
    int n = idx / K;
    int k = idx - n * K;
    Wt[idx] = (_Float16)W[k * N + n];
}

// ---------------------------------------------------------------------------
// Edge phase: msg = relu(x[src] + e0[a0]+e1[a1]+e2[a2]); atomic scatter to aggr[dst]
// One edge per wave iteration; lane owns 4 contiguous channels (float4).
// Embedding tables (13 rows x 128 = 6.6KB) staged in LDS.
// ---------------------------------------------------------------------------
#define EDGE_WAVES 8
#define EDGES_PER_WAVE 32
__global__ __launch_bounds__(256) void edge_kernel(const float* __restrict__ x,
                                                   const int* __restrict__ edge_index,
                                                   const int* __restrict__ edge_attr,
                                                   const float* __restrict__ e0,
                                                   const float* __restrict__ e1,
                                                   const float* __restrict__ e2,
                                                   float* __restrict__ aggr) {
    __shared__ float semb[1664];                    // 5*128 + 6*128 + 2*128
    for (int i = threadIdx.x; i < 640; i += 256) semb[i]        = e0[i];
    for (int i = threadIdx.x; i < 768; i += 256) semb[640 + i]  = e1[i];
    for (int i = threadIdx.x; i < 256; i += 256) semb[1408 + i] = e2[i];
    __syncthreads();

    const int lane = threadIdx.x & 31;
    const int wave = threadIdx.x >> 5;
    const int c    = lane * 4;                      // channel base for this lane
    const int base = (blockIdx.x * EDGE_WAVES + wave) * EDGES_PER_WAVE;

    for (int t = 0; t < EDGES_PER_WAVE; ++t) {
        const int e   = base + t;
        const int src = edge_index[e];
        const int dst = edge_index[N_EDGES + e];
        const int a0  = edge_attr[e * 3 + 0];
        const int a1  = edge_attr[e * 3 + 1];
        const int a2  = edge_attr[e * 3 + 2];

        float4 xv = *(const float4*)(x + (size_t)src * IN_DIM + c);
        float4 v0 = *(const float4*)(semb + a0 * 128 + c);
        float4 v1 = *(const float4*)(semb + 640 + a1 * 128 + c);
        float4 v2 = *(const float4*)(semb + 1408 + a2 * 128 + c);

        float m0 = fmaxf(xv.x + v0.x + v1.x + v2.x, 0.f);
        float m1 = fmaxf(xv.y + v0.y + v1.y + v2.y, 0.f);
        float m2 = fmaxf(xv.z + v0.z + v1.z + v2.z, 0.f);
        float m3 = fmaxf(xv.w + v0.w + v1.w + v2.w, 0.f);

        float* ap = aggr + (size_t)dst * IN_DIM + c;
        unsafeAtomicAdd(ap + 0, m0);
        unsafeAtomicAdd(ap + 1, m1);
        unsafeAtomicAdd(ap + 2, m2);
        unsafeAtomicAdd(ap + 3, m3);
    }
}

// ---------------------------------------------------------------------------
// Node update: hA = f16( (1+eps)*x + aggr )
// ---------------------------------------------------------------------------
__global__ __launch_bounds__(256) void node_update_kernel(const float* __restrict__ x,
                                                          const float* __restrict__ aggr,
                                                          const float* __restrict__ epsp,
                                                          _Float16* __restrict__ hA,
                                                          int n4) {
    int i = blockIdx.x * 256 + threadIdx.x;
    if (i >= n4) return;
    const float k = 1.f + *epsp;
    float4 xv = *(const float4*)(x + (size_t)i * 4);
    float4 av = *(const float4*)(aggr + (size_t)i * 4);
    H4 o;
    o.h[0] = (_Float16)(k * xv.x + av.x);
    o.h[1] = (_Float16)(k * xv.y + av.y);
    o.h[2] = (_Float16)(k * xv.z + av.z);
    o.h[3] = (_Float16)(k * xv.w + av.w);
    *(H4*)(hA + (size_t)i * 4) = o;
}

// ---------------------------------------------------------------------------
// GEMM1 (WMMA f16->f32) fused with bias + LayerNorm + ReLU, f16 output.
// Block: 512 threads = 16 waves; block owns a 16-row strip, wave w owns
// output cols [16w, 16w+16). K = 128 -> 4 x v_wmma_f32_16x16x32_f16.
// ---------------------------------------------------------------------------
#define LNP 260   // LDS row pitch (floats): 8*260 % 64 == 32 -> no bank conflicts
__global__ __launch_bounds__(512) void gemm1_ln_kernel(const _Float16* __restrict__ hA,
                                                       const _Float16* __restrict__ W1t,
                                                       const float* __restrict__ b1,
                                                       const float* __restrict__ ln_g,
                                                       const float* __restrict__ ln_b,
                                                       _Float16* __restrict__ h1) {
    __shared__ float sH[16 * LNP];
    const int wave = threadIdx.x >> 5;
    const int lane = threadIdx.x & 31;
    const int lo = lane & 15, hi = lane >> 4;
    const int m0 = blockIdx.x * 16;
    const int n0 = wave * 16;

    v8f c = {0.f, 0.f, 0.f, 0.f, 0.f, 0.f, 0.f, 0.f};
    const _Float16* arow = hA  + (size_t)(m0 + lo) * IN_DIM;
    const _Float16* bcol = W1t + (size_t)(n0 + lo) * IN_DIM;

#pragma unroll
    for (int ks = 0; ks < IN_DIM / 32; ++ks) {
        Frag a, b;
        const int ka = ks * 32 + hi * 8;            // A: K chunks [ka,ka+8) and [ka+16,ka+24)
        a.u[0] = *(const U4*)(arow + ka);
        a.u[1] = *(const U4*)(arow + ka + 16);
        const int kb = ks * 32 + hi * 8;            // B: K chunks [kb,kb+8) and [kb+16,kb+24)
        b.u[0] = *(const U4*)(bcol + kb);
        b.u[1] = *(const U4*)(bcol + kb + 16);
        c = __builtin_amdgcn_wmma_f32_16x16x32_f16(false, a.h, false, b.h,
                                                   (short)0, c, false, false);
    }

    const int n = n0 + lo;
    const float bias = b1[n];
#pragma unroll
    for (int r = 0; r < 8; ++r)
        sH[(r + 8 * hi) * LNP + n] = c[r] + bias;
    __syncthreads();

    // LayerNorm: wave handles row 'wave' of the strip; lane owns 8 columns.
    const float* rowp = sH + wave * LNP;
    float v[8], s = 0.f, s2 = 0.f;
#pragma unroll
    for (int j = 0; j < 8; ++j) {
        v[j] = rowp[lane + 32 * j];
        s += v[j];
        s2 += v[j] * v[j];
    }
#pragma unroll
    for (int off = 16; off > 0; off >>= 1) {
        s  += __shfl_xor(s, off, 32);
        s2 += __shfl_xor(s2, off, 32);
    }
    const float mean = s * (1.f / EMB_DIM);
    const float var  = s2 * (1.f / EMB_DIM) - mean * mean;
    const float rs   = rsqrtf(var + LN_EPS);

    _Float16* outp = h1 + (size_t)(m0 + wave) * EMB_DIM;
#pragma unroll
    for (int j = 0; j < 8; ++j) {
        const int col = lane + 32 * j;
        float y = (v[j] - mean) * rs * ln_g[col] + ln_b[col];
        outp[col] = (_Float16)fmaxf(y, 0.f);
    }
}

// ---------------------------------------------------------------------------
// GEMM2 (WMMA f16->f32) + bias, f32 output. K = 256 -> 8 WMMAs per tile.
// ---------------------------------------------------------------------------
__global__ __launch_bounds__(512) void gemm2_kernel(const _Float16* __restrict__ h1,
                                                    const _Float16* __restrict__ W2t,
                                                    const float* __restrict__ b2,
                                                    float* __restrict__ out) {
    const int wave = threadIdx.x >> 5;
    const int lane = threadIdx.x & 31;
    const int lo = lane & 15, hi = lane >> 4;
    const int m0 = blockIdx.x * 16;
    const int n0 = wave * 16;

    v8f c = {0.f, 0.f, 0.f, 0.f, 0.f, 0.f, 0.f, 0.f};
    const _Float16* arow = h1  + (size_t)(m0 + lo) * EMB_DIM;
    const _Float16* bcol = W2t + (size_t)(n0 + lo) * EMB_DIM;

#pragma unroll
    for (int ks = 0; ks < EMB_DIM / 32; ++ks) {
        Frag a, b;
        const int ka = ks * 32 + hi * 8;
        a.u[0] = *(const U4*)(arow + ka);
        a.u[1] = *(const U4*)(arow + ka + 16);
        const int kb = ks * 32 + hi * 8;
        b.u[0] = *(const U4*)(bcol + kb);
        b.u[1] = *(const U4*)(bcol + kb + 16);
        c = __builtin_amdgcn_wmma_f32_16x16x32_f16(false, a.h, false, b.h,
                                                   (short)0, c, false, false);
    }

    const int n = n0 + lo;
    const float bias = b2[n];
#pragma unroll
    for (int r = 0; r < 8; ++r) {
        const int m = m0 + r + 8 * hi;
        out[(size_t)m * EMB_DIM + n] = c[r] + bias;
    }
}

// ---------------------------------------------------------------------------
extern "C" void kernel_launch(void* const* d_in, const int* in_sizes, int n_in,
                              void* d_out, int out_size, void* d_ws, size_t ws_size,
                              hipStream_t stream) {
    const float* x          = (const float*)d_in[0];
    const int*   edge_index = (const int*)d_in[1];
    const int*   edge_attr  = (const int*)d_in[2];
    const float* bond_emb0  = (const float*)d_in[3];
    const float* bond_emb1  = (const float*)d_in[4];
    const float* bond_emb2  = (const float*)d_in[5];
    const float* epsp       = (const float*)d_in[6];
    const float* W1         = (const float*)d_in[7];
    const float* b1         = (const float*)d_in[8];
    const float* ln_g       = (const float*)d_in[9];
    const float* ln_b       = (const float*)d_in[10];
    const float* W2         = (const float*)d_in[11];
    const float* b2         = (const float*)d_in[12];
    float* out              = (float*)d_out;

    // Workspace layout (all 16B aligned)
    char* ws = (char*)d_ws;
    float*    aggr = (float*)ws;                              // 40000*128*4  = 20,480,000 B
    _Float16* hA   = (_Float16*)(ws + 20480000);              // 40000*128*2  = 10,240,000 B
    _Float16* h1   = (_Float16*)(ws + 30720000);              // 40000*256*2  = 20,480,000 B
    _Float16* W1t  = (_Float16*)(ws + 51200000);              // 256*128*2    =     65,536 B
    _Float16* W2t  = (_Float16*)(ws + 51265536);              // 256*256*2    =    131,072 B

    const int n4 = N_NODES * IN_DIM / 4;                      // 1,280,000

    zero_kernel<<<(n4 + 255) / 256, 256, 0, stream>>>((float4*)aggr, n4);

    wprep_kernel<<<(EMB_DIM * IN_DIM + 255) / 256, 256, 0, stream>>>(W1, W1t, IN_DIM, EMB_DIM);
    wprep_kernel<<<(EMB_DIM * EMB_DIM + 255) / 256, 256, 0, stream>>>(W2, W2t, EMB_DIM, EMB_DIM);

    edge_kernel<<<N_EDGES / (EDGE_WAVES * EDGES_PER_WAVE), 256, 0, stream>>>(
        x, edge_index, edge_attr, bond_emb0, bond_emb1, bond_emb2, aggr);

    node_update_kernel<<<(n4 + 255) / 256, 256, 0, stream>>>(x, aggr, epsp, hA, n4);

    gemm1_ln_kernel<<<N_NODES / 16, 512, 0, stream>>>(hA, W1t, b1, ln_g, ln_b, h1);

    gemm2_kernel<<<N_NODES / 16, 512, 0, stream>>>(h1, W2t, b2, out);
}
